// LinearLayerCRF_15040975470683
// MI455X (gfx1250) — matrix operand order repaired and checked by reading, not compiled, not run on hardware
//
#include <hip/hip_runtime.h>

typedef __attribute__((ext_vector_type(16))) _Float16 v16h;
typedef __attribute__((ext_vector_type(8)))  float    v8f;

#define BB     128
#define SS     4096
#define NTAG   48
#define EMBD   10
#define CHUNK  256
#define NCHUNK 16      // SS / CHUNK

// ---------------------------------------------------------------------------
// Kernel 1: fused emissions (WMMA) + Viterbi forward scan.
// One wave (32 threads) per batch row. Lane l owns tag jA=l and, for l<16,
// tag jB=l+32 (48 tags total). trans columns live in registers (unrolled).
// ---------------------------------------------------------------------------
__global__ __launch_bounds__(32) void crf_fwd(
    const int*   __restrict__ token_ids,   // [BB, SS]
    const float* __restrict__ emb,         // [VOCAB, EMBD]
    const float* __restrict__ W,           // [NTAG, EMBD]
    const float* __restrict__ bvec,        // [NTAG]
    const float* __restrict__ startt,      // [NTAG]
    const float* __restrict__ endt,        // [NTAG]
    const float* __restrict__ trans,       // [NTAG, NTAG]
    unsigned char* __restrict__ bp,        // [BB, SS, NTAG] backpointers
    int*         __restrict__ last_tag)    // [BB]
{
    __shared__ float emled[16][NTAG];   // one 16-timestep emission tile
    __shared__ float sc[NTAG];          // broadcast of previous scores
    __shared__ float fin[NTAG];

    const int b    = blockIdx.x;
    const int lane = threadIdx.x;
    const int jA   = lane;
    const int jB   = lane + 32;
    const int jBc  = (lane < 16) ? jB : 0;   // clamped for safe loads

    // --- register-resident transition columns -----------------------------
    float trA[NTAG], trB[NTAG];
#pragma unroll
    for (int i = 0; i < NTAG; ++i) {
        trA[i] = trans[i * NTAG + jA];
        trB[i] = trans[i * NTAG + jBc];
    }
    const float stA = startt[jA],  stB = startt[jBc];
    const float enA = endt[jA],    enB = endt[jBc];

    // --- W packed as WMMA B-tiles (32x16 f16, K=EMBD zero-padded) ---------
    // B in this code's contiguous k order (k=16h+s): lanes 0-15 hold K=0..15 in halves 0..15 for column
    // n = lane&15; lanes 16-31 hold K=16..31 (all zero here since K>=10 is 0).
    const int n = lane & 15;
    v16h bm0 = {}, bm1 = {}, bm2 = {};
    if (lane < 16) {
#pragma unroll
        for (int k = 0; k < EMBD; ++k) {
            bm0[k] = (_Float16)W[(0 * 16 + n) * EMBD + k];
            bm1[k] = (_Float16)W[(1 * 16 + n) * EMBD + k];
            bm2[k] = (_Float16)W[(2 * 16 + n) * EMBD + k];
        }
    }
    const float bias0 = bvec[n], bias1 = bvec[16 + n], bias2 = bvec[32 + n];

    const int*     tok   = token_ids + b * SS;
    unsigned char* bprow = bp + (size_t)b * SS * NTAG;

    float scA = 0.f, scB = 0.f;

    for (int tc = 0; tc < SS; tc += 16) {
        // ---- emissions for timesteps [tc, tc+16) via 3 WMMAs -------------
        // A (16x32 f16), contiguous k order as B: lane%16 = row m; lanes<16 halves0-15 = K0..15
        // (only K0..K9 nonzero); lanes>=16 halves0-15 = K16..31 (zero).
        const int m   = lane & 15;
        const int tid = tok[tc + m];
        const float* er = emb + (size_t)tid * EMBD;
        v16h a = {};
        if (lane < 16) {
#pragma unroll
            for (int k = 0; k < 8; ++k) a[k] = (_Float16)er[k];
            a[8] = (_Float16)er[8];
            a[9] = (_Float16)er[9];
        }

        v8f c0 = {}, c1 = {}, c2 = {};
        c0 = __builtin_amdgcn_wmma_f32_16x16x32_f16(false, a, false, bm0,
                                                    (short)0, c0, false, false);
        c1 = __builtin_amdgcn_wmma_f32_16x16x32_f16(false, a, false, bm1,
                                                    (short)0, c1, false, false);
        c2 = __builtin_amdgcn_wmma_f32_16x16x32_f16(false, a, false, bm2,
                                                    (short)0, c2, false, false);

        // D layout: vgpr r -> row = r + (lane>=16 ? 8 : 0), col = lane&15
        const int rb = (lane >> 4) * 8;
#pragma unroll
        for (int r = 0; r < 8; ++r) {
            emled[rb + r][n]      = c0[r] + bias0;
            emled[rb + r][16 + n] = c1[r] + bias1;
            emled[rb + r][32 + n] = c2[r] + bias2;
        }
        __syncthreads();   // single wave -> just waitcnts

        // ---- 16 sequential Viterbi steps ---------------------------------
        for (int st = 0; st < 16; ++st) {
            const int t = tc + st;
            const float emA = emled[st][jA];
            const float emB = emled[st][jBc];

            if (t == 0) {
                scA = stA + emA;
                scB = stB + emB;
            } else {
                float bestA = sc[0] + trA[0]; int bpA = 0;
                float bestB = sc[0] + trB[0]; int bpB = 0;
#pragma unroll
                for (int i = 1; i < NTAG; ++i) {
                    const float s  = sc[i];         // broadcast read
                    const float vA = s + trA[i];
                    const bool  gA = vA > bestA;    // strict > == first-max tie
                    bestA = gA ? vA : bestA;
                    bpA   = gA ? i  : bpA;
                    const float vB = s + trB[i];
                    const bool  gB = vB > bestB;
                    bestB = gB ? vB : bestB;
                    bpB   = gB ? i  : bpB;
                }
                scA = bestA + emA;
                scB = bestB + emB;
                __syncthreads();                    // readers done before write
                bprow[t * NTAG + jA] = (unsigned char)bpA;
                if (lane < 16) bprow[t * NTAG + jB] = (unsigned char)bpB;
            }
            sc[jA] = scA;
            if (lane < 16) sc[jB] = scB;
            __syncthreads();
        }
    }

    // ---- final: argmax_j(score[j] + end_trans[j]) ------------------------
    fin[jA] = scA + enA;
    if (lane < 16) fin[jB] = scB + enB;
    __syncthreads();
    if (lane == 0) {
        float best = fin[0]; int bi = 0;
        for (int j = 1; j < NTAG; ++j) {
            if (fin[j] > best) { best = fin[j]; bi = j; }
        }
        last_tag[b] = bi;
    }
}

// ---------------------------------------------------------------------------
// Kernel 2: per-chunk hypothesis maps. For chunk c, walking all 48 possible
// end tags backward through the chunk gives map_c[h] = tag at (lo_c - 1).
// ---------------------------------------------------------------------------
__global__ __launch_bounds__(64) void crf_bt_maps(
    const unsigned char* __restrict__ bp,
    unsigned char*       __restrict__ cmap)   // [BB, NCHUNK, NTAG]
{
    const int bc = blockIdx.x;
    const int b  = bc >> 4;          // / NCHUNK
    const int c  = bc & (NCHUNK - 1);
    const int h  = threadIdx.x;
    if (h >= NTAG) return;

    const int lo   = c * CHUNK;
    const int hi   = lo + CHUNK - 1;
    const int tmin = (c == 0) ? 1 : lo;   // bp[0] does not exist
    const unsigned char* row = bp + (size_t)b * SS * NTAG;

    int cur = h;
    for (int t = hi; t >= tmin; --t)
        cur = row[t * NTAG + cur];

    cmap[((size_t)b * NCHUNK + c) * NTAG + h] = (unsigned char)cur;
}

// ---------------------------------------------------------------------------
// Kernel 3: stitch chunk maps into boundary tags, then 16 threads re-walk
// their chunk in parallel emitting the final tag sequence.
// ---------------------------------------------------------------------------
__global__ __launch_bounds__(64) void crf_bt_emit(
    const unsigned char* __restrict__ bp,
    const unsigned char* __restrict__ cmap,
    const int*           __restrict__ last_tag,
    int*                 __restrict__ out)    // [BB, SS]
{
    __shared__ int te[NCHUNK];   // tag at hi of each chunk
    const int b = blockIdx.x;

    if (threadIdx.x == 0) {
        int e = last_tag[b];
        te[NCHUNK - 1] = e;
        for (int c = NCHUNK - 1; c >= 1; --c) {
            e = cmap[((size_t)b * NCHUNK + c) * NTAG + e];
            te[c - 1] = e;
        }
    }
    __syncthreads();

    const int c = threadIdx.x;
    if (c < NCHUNK) {
        const int lo = c * CHUNK;
        const int hi = lo + CHUNK - 1;
        const unsigned char* row = bp + (size_t)b * SS * NTAG;
        int* ob = out + (size_t)b * SS;

        int E = te[c];
        ob[hi] = E;
        for (int t = hi; t > lo; --t) {
            E = row[t * NTAG + E];
            ob[t - 1] = E;
        }
    }
}

// ---------------------------------------------------------------------------
extern "C" void kernel_launch(void* const* d_in, const int* in_sizes, int n_in,
                              void* d_out, int out_size, void* d_ws, size_t ws_size,
                              hipStream_t stream) {
    const int*   token_ids = (const int*)  d_in[0];
    const float* emb       = (const float*)d_in[1];
    const float* W         = (const float*)d_in[2];
    const float* bvec      = (const float*)d_in[3];
    const float* startt    = (const float*)d_in[4];
    const float* endt      = (const float*)d_in[5];
    const float* trans     = (const float*)d_in[6];
    int* out = (int*)d_out;

    // workspace layout
    const size_t bpBytes = (size_t)BB * SS * NTAG;        // ~25.2 MB
    unsigned char* bp       = (unsigned char*)d_ws;
    int*           last_tag = (int*)((char*)d_ws + bpBytes);
    unsigned char* cmap     = (unsigned char*)d_ws + bpBytes + 1024;

    crf_fwd<<<BB, 32, 0, stream>>>(token_ids, emb, W, bvec, startt, endt,
                                   trans, bp, last_tag);
    crf_bt_maps<<<BB * NCHUNK, 64, 0, stream>>>(bp, cmap);
    crf_bt_emit<<<BB, 64, 0, stream>>>(bp, cmap, last_tag, out);
}
